// MarkovRouteChoice_26147760898132
// MI455X (gfx1250) — matrix operand order repaired and checked by reading, not compiled, not run on hardware
//
#include <hip/hip_runtime.h>
#include <hip/hip_bf16.h>
#include <math.h>

// ---------------------------------------------------------------------------
// MarkovRouteChoice on MI455X (gfx1250, wave32, WMMA)
//
// outputs (concat): log_r [E] | log_z [N] | edge_probs [E]
// workspace: packed-W1 frags (4KB) | r[E] | z ping-pong 2*[N]
// ---------------------------------------------------------------------------

typedef __attribute__((ext_vector_type(16))) _Float16 v16h;
typedef __attribute__((ext_vector_type(8)))  float    v8f;

#define F_IN    16
#define HIDDEN  64
#define ITERS   50
#define COST_OFFSET 4.0f

#define WAVES_PER_BLOCK  8
#define EDGES_PER_WAVE   16
#define EDGES_PER_BLOCK  (WAVES_PER_BLOCK * EDGES_PER_WAVE)   // 128
#define HSTRIDE          72                                   // 16B-aligned rows

// -------------------- Kernel 0: pack W1 into B-fragment layout -------------
// One block, 128 threads = 4 tiles x 32 lanes. For tile t, lane l:
//   halfs[i] = (i < 8) ? f16(W1[8*(l/16) + i][16t + l%16]) : 0      (i = 0..15)
// This is exactly the per-lane v16h operand of v_wmma_f32_16x16x32_f16
// (elements 0-7 carry K=8h..8h+7; elements 8-15 carry the zero K>=16 pad).
__global__ void mrc_pack_w1_k(const float* __restrict__ W1,   // [16,64]
                              _Float16* __restrict__ w1f)     // [4*32*16]
{
    const int tid  = threadIdx.x;          // 0..127
    const int t    = tid >> 5;             // tile 0..3
    const int lane = tid & 31;
    const int half = lane >> 4;
    const int lq   = lane & 15;
    const int n    = t * 16 + lq;
    _Float16* out = w1f + (t * 32 + lane) * 16;
    #pragma unroll
    for (int i = 0; i < 16; ++i) {
        float w = W1[(half * 8 + (i & 7)) * HIDDEN + n];      // unconditional load (n always valid)
        out[i] = (i < 8) ? (_Float16)w : (_Float16)0.0f;
    }
}

// -------------------- Kernel 1: edge encoder (WMMA) ------------------------
__global__ void mrc_encoder(const float* __restrict__ x,     // [E,16]
                            const _Float16* __restrict__ w1f,// packed B frags
                            const float* __restrict__ b1,    // [64]
                            const float* __restrict__ W2,    // [64]
                            const float* __restrict__ b2,    // [1]
                            float* __restrict__ r_out,       // [E]  (ws)
                            float* __restrict__ logr_out,    // [E]  (d_out)
                            long long E)
{
    __shared__ float h_sh[WAVES_PER_BLOCK][16][HSTRIDE];     // rows 16B-aligned
    __shared__ float w2_sh[HIDDEN];

    const int tid  = threadIdx.x;
    const int wave = tid >> 5;          // wave32
    const int lane = tid & 31;
    const int half = lane >> 4;
    const int lq   = lane & 15;

    if (tid < HIDDEN) w2_sh[tid] = W2[tid];

    const long long ebase = (long long)blockIdx.x * EDGES_PER_BLOCK
                          + (long long)wave * EDGES_PER_WAVE;

    // ---- B fragments: one clean 32B vector load per tile (no branches) ----
    const v16h* __restrict__ bf = (const v16h*)w1f;
    v16h b0   = bf[0 * 32 + lane];
    v16h b1f_ = bf[1 * 32 + lane];
    v16h b2f_ = bf[2 * 32 + lane];
    v16h b3f_ = bf[3 * 32 + lane];

    // ---- A fragment: 16x32 f16, K>=16 zero pad. Lane reads 8 contiguous
    // 32B-aligned floats (row*64B + half*32B) as two float4. ----
    v16h a;
    {
        long long row = ebase + lq;
        if (row >= E) row = E - 1;               // clamp; results discarded
        const float4* xp = (const float4*)(x + row * F_IN + half * 8);
        float4 x0 = xp[0];
        float4 x1 = xp[1];
        a[0] = (_Float16)x0.x; a[1] = (_Float16)x0.y;
        a[2] = (_Float16)x0.z; a[3] = (_Float16)x0.w;
        a[4] = (_Float16)x1.x; a[5] = (_Float16)x1.y;
        a[6] = (_Float16)x1.z; a[7] = (_Float16)x1.w;
        #pragma unroll
        for (int i = 8; i < 16; ++i) a[i] = (_Float16)0.0f;
    }

    // ---- 4 N-tiles of 16 columns ----
    v8f acc[4];
    {
        v8f c = {};
        acc[0] = __builtin_amdgcn_wmma_f32_16x16x32_f16(false, a, false, b0,  (short)0, c, false, false);
        acc[1] = __builtin_amdgcn_wmma_f32_16x16x32_f16(false, a, false, b1f_,(short)0, c, false, false);
        acc[2] = __builtin_amdgcn_wmma_f32_16x16x32_f16(false, a, false, b2f_,(short)0, c, false, false);
        acc[3] = __builtin_amdgcn_wmma_f32_16x16x32_f16(false, a, false, b3f_,(short)0, c, false, false);
    }

    // ---- h = relu(acc + b1) -> LDS. C/D layout: vgpr v, lane l holds
    // row M = v + 8*(l/16), col N = (l%16) within 16-col tile t. ----
    #pragma unroll
    for (int t = 0; t < 4; ++t) {
        const int col  = t * 16 + lq;
        const float bv = b1[col];
        #pragma unroll
        for (int v = 0; v < 8; ++v) {
            const int rowm = v + 8 * half;
            float hv = acc[t][v] + bv;
            h_sh[wave][rowm][col] = hv > 0.0f ? hv : 0.0f;
        }
    }
    __syncthreads();

    // ---- second layer: one edge per thread (threads 0..127) ----
    if (tid < EDGES_PER_BLOCK) {
        const long long e = (long long)blockIdx.x * EDGES_PER_BLOCK + tid;
        if (e < E) {
            const float4* hrow = (const float4*)&h_sh[tid >> 4][tid & 15][0];
            const float4* w2v  = (const float4*)w2_sh;
            float acc2 = b2[0];
            #pragma unroll
            for (int q = 0; q < HIDDEN / 4; ++q) {
                float4 h4 = hrow[q];
                float4 w4 = w2v[q];
                acc2 += h4.x * w4.x + h4.y * w4.y + h4.z * w4.z + h4.w * w4.w;
            }
            // numerically stable softplus: max(x,0) + log1p(exp(-|x|))
            float sp   = fmaxf(acc2, 0.0f) + log1pf(expf(-fabsf(acc2)));
            float cost = sp + COST_OFFSET;
            logr_out[e] = -cost;
            r_out[e]    = expf(-cost);
        }
    }
}

// -------------------- Kernel 2: z := b (sink mask) -------------------------
__global__ void mrc_vi_init(const unsigned char* __restrict__ mask,  // [N] bool
                            float* __restrict__ z, int N)
{
    int i = blockIdx.x * blockDim.x + threadIdx.x;
    if (i < N) z[i] = mask[i] ? 1.0f : 0.0f;
}

// -------------------- Kernel 3: value-iteration step -----------------------
// znew[src[e]] += r[e] * z[dst[e]]   (znew pre-initialized to b)
__global__ void mrc_vi_step(const int* __restrict__ src,
                            const int* __restrict__ dst,
                            const float* __restrict__ r,
                            const float* __restrict__ z,
                            float* __restrict__ znew, int E)
{
    int e = blockIdx.x * blockDim.x + threadIdx.x;
    if (e < E) {
        float msg = r[e] * z[dst[e]];          // gather (L2-resident z)
        atomicAdd(&znew[src[e]], msg);         // global_atomic_add_f32
    }
}

// -------------------- Kernel 4: finalize -----------------------------------
__global__ void mrc_finalize(const int* __restrict__ src,
                             const int* __restrict__ dst,
                             const float* __restrict__ r,
                             const float* __restrict__ z,
                             float* __restrict__ out_logz,   // [N]
                             float* __restrict__ out_probs,  // [E]
                             int E, int N)
{
    int i = blockIdx.x * blockDim.x + threadIdx.x;
    if (i < N) out_logz[i] = logf(z[i]);
    if (i < E) out_probs[i] = r[i] * z[dst[i]] / z[src[i]];
}

// ---------------------------------------------------------------------------
extern "C" void kernel_launch(void* const* d_in, const int* in_sizes, int n_in,
                              void* d_out, int out_size, void* d_ws, size_t ws_size,
                              hipStream_t stream)
{
    // inputs (setup_inputs order):
    // 0: edge_index [2,E] int32   1: edge_feats [E,16] f32
    // 2: sink_node_mask [N] bool  3: W1 [16,64]  4: b1 [64]  5: W2 [64,1]  6: b2 [1]
    const int E = in_sizes[0] / 2;
    const int N = in_sizes[2];

    const int*   src  = (const int*)d_in[0];
    const int*   dst  = src + E;
    const float* x    = (const float*)d_in[1];
    const unsigned char* mask = (const unsigned char*)d_in[2];
    const float* W1   = (const float*)d_in[3];
    const float* b1   = (const float*)d_in[4];
    const float* W2   = (const float*)d_in[5];
    const float* b2   = (const float*)d_in[6];

    float* out_logr  = (float*)d_out;              // [E]
    float* out_logz  = out_logr + E;               // [N]
    float* out_probs = out_logz + N;               // [E]

    // workspace layout (d_ws is >=256B aligned):
    _Float16* w1f = (_Float16*)d_ws;               // 4*32*16 halfs = 4KB
    float* r  = (float*)((char*)d_ws + 4096);      // [E]
    float* z0 = r + E;                             // [N]
    float* z1 = z0 + N;                            // [N]

    // 0) pack W1 into per-lane WMMA B-fragment layout (one tiny block)
    mrc_pack_w1_k<<<1, 128, 0, stream>>>(W1, w1f);

    // 1) encoder: r, log_r (WMMA path)
    {
        dim3 grid((E + EDGES_PER_BLOCK - 1) / EDGES_PER_BLOCK);
        dim3 block(WAVES_PER_BLOCK * 32);
        mrc_encoder<<<grid, block, 0, stream>>>(x, w1f, b1, W2, b2,
                                                r, out_logr, (long long)E);
    }

    // 2) value iteration: z = b; repeat 50x { znew = b; znew += scatter(r*z[dst]) }
    const int TB = 256;
    dim3 gN((N + TB - 1) / TB), gE((E + TB - 1) / TB), blk(TB);
    mrc_vi_init<<<gN, blk, 0, stream>>>(mask, z0, N);
    float* zcur = z0;
    float* znxt = z1;
    for (int it = 0; it < ITERS; ++it) {
        mrc_vi_init<<<gN, blk, 0, stream>>>(mask, znxt, N);
        mrc_vi_step<<<gE, blk, 0, stream>>>(src, dst, r, zcur, znxt, E);
        float* t = zcur; zcur = znxt; znxt = t;
    }

    // 3) finalize: log_z and edge_probs
    {
        int M = (E > N) ? E : N;
        dim3 g((M + TB - 1) / TB);
        mrc_finalize<<<g, blk, 0, stream>>>(src, dst, r, zcur,
                                            out_logz, out_probs, E, N);
    }
}
